// MLA_29867202576994
// MI455X (gfx1250) — matrix operand order repaired and checked by reading, not compiled, not run on hardware
//
#include <hip/hip_runtime.h>
#include <hip/hip_bf16.h>
#include <math.h>

// ---------------- types ----------------
typedef __attribute__((ext_vector_type(16))) __bf16        v16bf;
typedef __attribute__((ext_vector_type(8)))  float         v8f;
typedef __attribute__((ext_vector_type(4)))  unsigned int  v4u;

union FragBF { v16bf v; v4u u[2]; };

__device__ __forceinline__ unsigned short f2bf(float f) {
    unsigned int u = __float_as_uint(f);
    u += 0x7FFFu + ((u >> 16) & 1u);   // round-to-nearest-even
    return (unsigned short)(u >> 16);
}

__device__ __forceinline__ void storeC(float* p, float v)          { *p = v; }
__device__ __forceinline__ void storeC(unsigned short* p, float v) { *p = f2bf(v); }

// ---------------- WMMA NT-GEMM (contiguous K for both operands) ----------------
// C(m,n) = alpha * sum_k A[m*sAm + k] * B[n*sBn + k]  [+ Add(m,n)]
// C element (m,n) at C[m*sCm + n*sCn].  blockIdx.z batches via sAh/sBh/sCh.
// Block tile 128x128, 8 waves (4 along M x 2 along N), wave tile 32x64, BK=32.
// Software-pipelined: double-buffered LDS, K-loop unrolled x2 (static ping-pong).
// Requires K/BK even and >= 2 (true for all launches here: 6, 16, 64).
#define BM 128
#define BN 128
#define BK 32
#define LDSTR 40   // halves per LDS row (80B, 16B aligned, bank-skewed)

template <typename OutT, bool HasAdd>
__global__ __launch_bounds__(256) void gemm_nt(
    const unsigned short* __restrict__ A, long sAm, long sAh,
    const unsigned short* __restrict__ B, long sBn, long sBh,
    OutT* __restrict__ C, long sCm, long sCn, long sCh,
    const float* __restrict__ Add, long sAddM,
    int N, int K, float alpha)
{
    __shared__ __align__(16) unsigned short lA[2][BM * LDSTR];
    __shared__ __align__(16) unsigned short lB[2][BN * LDSTR];

    const int tid = threadIdx.x;
    const int bm  = blockIdx.y * BM;
    const int bn  = blockIdx.x * BN;
    const long hh = blockIdx.z;
    A += hh * sAh; B += hh * sBh; C += hh * sCh;

    const int wave = tid >> 5, lane = tid & 31;
    const int wm = (wave & 3) * 32;   // 4 waves along M
    const int wn = (wave >> 2) * 64;  // 2 waves along N

    // wave32 WMMA fragment lane mapping (16-bit A 16x32 / B 32x16, ISA 7.12.2)
    const int aRow = lane & 15;
    const int aK   = (lane >> 4) * 8;    // A: chunks [aK..aK+7], [aK+16..aK+23]
    const int bCol = lane & 15;
    const int bK   = (lane >> 4) * 8;    // B: chunks [bK..bK+7], [bK+16..bK+23]

    // staging: 256 threads, 2 threads/row, 16 halves (2 x b128) each
    const int arow = tid >> 1, akof = (tid & 1) * 16;
    const int brow = tid >> 1, bkof = (tid & 1) * 16;
    int bRowG = bn + brow; if (bRowG > N - 1) bRowG = N - 1;  // clamp for N=64 launches

    const unsigned short* aPtr = A + (long)(bm + arow) * sAm + akof;
    const unsigned short* bPtr = B + (long)bRowG * sBn + bkof;
    unsigned short* const sa0 = &lA[0][arow * LDSTR + akof];
    unsigned short* const sb0 = &lB[0][brow * LDSTR + bkof];
    unsigned short* const sa1 = &lA[1][arow * LDSTR + akof];
    unsigned short* const sb1 = &lB[1][brow * LDSTR + bkof];

    v8f zero = {0.f,0.f,0.f,0.f,0.f,0.f,0.f,0.f};
    v8f acc[2][4] = {{zero, zero, zero, zero}, {zero, zero, zero, zero}};

    auto computeTile = [&](const unsigned short* la, const unsigned short* lb) {
        FragBF fa[2], fb[4];
        #pragma unroll
        for (int mi = 0; mi < 2; ++mi) {
            const unsigned short* p = &la[(wm + mi * 16 + aRow) * LDSTR + aK];
            fa[mi].u[0] = *(const v4u*)p;
            fa[mi].u[1] = *(const v4u*)(p + 16);
        }
        #pragma unroll
        for (int ni = 0; ni < 4; ++ni) {
            const unsigned short* p = &lb[(wn + ni * 16 + bCol) * LDSTR + bK];
            fb[ni].u[0] = *(const v4u*)p;
            fb[ni].u[1] = *(const v4u*)(p + 16);
        }
        #pragma unroll
        for (int mi = 0; mi < 2; ++mi)
            #pragma unroll
            for (int ni = 0; ni < 4; ++ni)
                acc[mi][ni] = __builtin_amdgcn_wmma_f32_16x16x32_bf16(
                    false, fa[mi].v, false, fb[ni].v, (short)0, acc[mi][ni], false, false);
    };

    // prologue: tile 0 -> buf0, tile 1 -> regs
    v4u ra0 = *(const v4u*)aPtr, ra1 = *(const v4u*)(aPtr + 8);
    v4u rb0 = *(const v4u*)bPtr, rb1 = *(const v4u*)(bPtr + 8);
    aPtr += BK; bPtr += BK;
    *(v4u*)sa0 = ra0; *(v4u*)(sa0 + 8) = ra1;
    *(v4u*)sb0 = rb0; *(v4u*)(sb0 + 8) = rb1;
    ra0 = *(const v4u*)aPtr; ra1 = *(const v4u*)(aPtr + 8);
    rb0 = *(const v4u*)bPtr; rb1 = *(const v4u*)(bPtr + 8);
    aPtr += BK; bPtr += BK;
    __syncthreads();

    const int nT = K / BK;   // even, >= 2
    for (int it = 0; it < nT; it += 2) {
        const bool more = (it + 2) < nT;
        // stage tile(it+1) -> buf1 (regs already resident)
        *(v4u*)sa1 = ra0; *(v4u*)(sa1 + 8) = ra1;
        *(v4u*)sb1 = rb0; *(v4u*)(sb1 + 8) = rb1;
        v4u na0, na1, nb0, nb1;
        if (more) {   // issue tile(it+2) loads + L2 prefetch ahead of the WMMA chain
            na0 = *(const v4u*)aPtr; na1 = *(const v4u*)(aPtr + 8);
            nb0 = *(const v4u*)bPtr; nb1 = *(const v4u*)(bPtr + 8);
            __builtin_prefetch(aPtr + 2 * BK, 0, 3);
            __builtin_prefetch(bPtr + 2 * BK, 0, 3);
            aPtr += BK; bPtr += BK;
        }
        computeTile(lA[0], lB[0]);      // tile it
        __syncthreads();                // buf1 visible; buf0 reads done
        if (more) {
            *(v4u*)sa0 = na0; *(v4u*)(sa0 + 8) = na1;
            *(v4u*)sb0 = nb0; *(v4u*)(sb0 + 8) = nb1;
            ra0 = *(const v4u*)aPtr; ra1 = *(const v4u*)(aPtr + 8);
            rb0 = *(const v4u*)bPtr; rb1 = *(const v4u*)(bPtr + 8);
            aPtr += BK; bPtr += BK;
        }
        computeTile(lA[1], lB[1]);      // tile it+1
        __syncthreads();                // buf0 visible; buf1 reads done
    }

    // epilogue: 32-bit C/D layout — lane<16: N=lane, M=j ; lane>=16: N=lane-16, M=8+j
    const int ncol  = lane & 15;
    const int mbase = (lane >> 4) * 8;
    #pragma unroll
    for (int mi = 0; mi < 2; ++mi)
        #pragma unroll
        for (int ni = 0; ni < 4; ++ni) {
            const int n = bn + wn + ni * 16 + ncol;
            if (n < N) {
                #pragma unroll
                for (int j = 0; j < 8; ++j) {
                    const int m = bm + wm + mi * 16 + mbase + j;
                    float v = alpha * acc[mi][ni][j];
                    if (HasAdd) v += Add[(long)m * sAddM + n];
                    storeC(C + (long)m * sCm + (long)n * sCn, v);
                }
            }
        }
}

// ---------------- elementwise kernels ----------------
__global__ __launch_bounds__(256) void f32_to_bf16_k(const float* __restrict__ in,
                                                     unsigned short* __restrict__ out, int n) {
    int i = blockIdx.x * 256 + threadIdx.x;
    if (i < n) out[i] = f2bf(in[i]);
}

// q_r rope: qr_f32 [NH][T][64] -> qfull [NH][T][192] cols 128..191 (bf16)
__global__ __launch_bounds__(256) void rope_q_k(const float* __restrict__ qr,
                                                const float* __restrict__ fr,
                                                unsigned short* __restrict__ qfull) {
    int i = blockIdx.x * 256 + threadIdx.x;      // < 16*2048*32
    if (i >= 16 * 2048 * 32) return;
    int j = i & 31, t = (i >> 5) & 2047, n = i >> 16;
    float a = fr[t * 32 + j];
    size_t src = ((size_t)n * 2048 + t) * 64;
    float re = qr[src + j], im = qr[src + 32 + j];
    float c = cosf(a), s = sinf(a);
    size_t base = ((size_t)n * 2048 + t) * 192;
    qfull[base + 128 + j] = f2bf(re * c - im * s);
    qfull[base + 160 + j] = f2bf(re * s + im * c);
}

// k_r rope + 1/NH scale, broadcast to all heads' kfull cols 128..191
__global__ __launch_bounds__(256) void rope_k_k(const float* __restrict__ kr,
                                                const float* __restrict__ fr,
                                                unsigned short* __restrict__ kfull) {
    int i = blockIdx.x * 256 + threadIdx.x;      // < 2048*32
    if (i >= 2048 * 32) return;
    int j = i & 31, t = i >> 5;
    float a = fr[t * 32 + j];
    float re = kr[(size_t)t * 64 + j], im = kr[(size_t)t * 64 + 32 + j];
    float c = cosf(a), s = sinf(a);
    unsigned short zr = f2bf((re * c - im * s) * 0.0625f);
    unsigned short zi = f2bf((re * s + im * c) * 0.0625f);
    #pragma unroll
    for (int n = 0; n < 16; ++n) {
        size_t base = ((size_t)n * 2048 + t) * 192;
        kfull[base + 128 + j] = zr;
        kfull[base + 160 + j] = zi;
    }
}

// row softmax fp32 -> bf16, written TRANSPOSED: S[col*T + row]
__global__ __launch_bounds__(256) void softmax_k(const float* __restrict__ L,
                                                 unsigned short* __restrict__ S, int cols) {
    __shared__ float red[256];
    int row = blockIdx.x, tid = threadIdx.x;
    const float* src = L + (size_t)row * cols;
    float m = -3.4e38f;
    for (int c = tid; c < cols; c += 256) m = fmaxf(m, src[c]);
    red[tid] = m; __syncthreads();
    for (int s = 128; s > 0; s >>= 1) { if (tid < s) red[tid] = fmaxf(red[tid], red[tid + s]); __syncthreads(); }
    float rmax = red[0]; __syncthreads();
    float sum = 0.f;
    for (int c = tid; c < cols; c += 256) sum += __expf(src[c] - rmax);
    red[tid] = sum; __syncthreads();
    for (int s = 128; s > 0; s >>= 1) { if (tid < s) red[tid] += red[tid + s]; __syncthreads(); }
    float inv = 1.0f / red[0];
    for (int c = tid; c < cols; c += 256)
        S[(size_t)c * cols + row] = f2bf(__expf(src[c] - rmax) * inv);
}

// ---------------- host launch ----------------
extern "C" void kernel_launch(void* const* d_in, const int* in_sizes, int n_in,
                              void* d_out, int out_size, void* d_ws, size_t ws_size,
                              hipStream_t stream) {
    (void)in_sizes; (void)n_in; (void)out_size; (void)ws_size;
    const float* h     = (const float*)d_in[0];
    const float* freqs = (const float*)d_in[1];
    const float* mask  = (const float*)d_in[2];
    const float* w_dkv = (const float*)d_in[3];
    const float* w_uk  = (const float*)d_in[4];
    const float* w_uv  = (const float*)d_in[5];
    const float* w_dq  = (const float*)d_in[6];
    const float* w_uq  = (const float*)d_in[7];
    const float* w_qr  = (const float*)d_in[8];
    const float* w_kr  = (const float*)d_in[9];
    const float* w_o   = (const float*)d_in[10];
    float* out = (float*)d_out;

    constexpr int T = 2048, D = 2048, DC = 512, NH = 16, DH = 128, DRH = 64, QD = DH + DRH;

    char* ws = (char*)d_ws;
    size_t off = 0;
    auto alloc = [&](size_t bytes) -> char* {
        char* p = ws + off; off = (off + bytes + 255) & ~(size_t)255; return p;
    };
    unsigned short* h_bf    = (unsigned short*)alloc((size_t)T * D * 2);
    unsigned short* wdkv_bf = (unsigned short*)alloc((size_t)DC * D * 2);
    unsigned short* wdq_bf  = (unsigned short*)alloc((size_t)DC * D * 2);
    unsigned short* wkr_bf  = (unsigned short*)alloc((size_t)DRH * D * 2);
    unsigned short* wuk_bf  = (unsigned short*)alloc((size_t)DH * NH * DC * 2);
    unsigned short* wuv_bf  = (unsigned short*)alloc((size_t)DH * NH * DC * 2);
    unsigned short* wuq_bf  = (unsigned short*)alloc((size_t)DH * NH * DC * 2);
    unsigned short* wqr_bf  = (unsigned short*)alloc((size_t)DRH * NH * DC * 2);
    unsigned short* wo_bf   = (unsigned short*)alloc((size_t)D * DH * NH * 2);
    unsigned short* ckv     = (unsigned short*)alloc((size_t)T * DC * 2);
    unsigned short* cq      = (unsigned short*)alloc((size_t)T * DC * 2);
    unsigned short* qfull   = (unsigned short*)alloc((size_t)NH * T * QD * 2);
    unsigned short* kfull   = (unsigned short*)alloc((size_t)NH * T * QD * 2);
    unsigned short* vT      = (unsigned short*)alloc((size_t)NH * DH * T * 2);   // [n][e][t]
    float*          qr_f32  = (float*)alloc((size_t)NH * T * DRH * 4);
    float*          kr_f32  = (float*)alloc((size_t)T * DRH * 4);
    float*          logits  = (float*)alloc((size_t)T * T * 4);          // per-head, reused
    unsigned short* scoresT = (unsigned short*)alloc((size_t)T * T * 2); // per-head, reused, [l][t]
    unsigned short* att     = (unsigned short*)alloc((size_t)T * DH * NH * 2);   // [l][e*16+n]

    dim3 blk(256);
    auto cvt = [&](const float* s, unsigned short* d, int n) {
        f32_to_bf16_k<<<(n + 255) / 256, 256, 0, stream>>>(s, d, n);
    };
    cvt(h, h_bf, T * D);
    cvt(w_dkv, wdkv_bf, DC * D);
    cvt(w_dq, wdq_bf, DC * D);
    cvt(w_kr, wkr_bf, DRH * D);
    cvt(w_uk, wuk_bf, DH * NH * DC);
    cvt(w_uv, wuv_bf, DH * NH * DC);
    cvt(w_uq, wuq_bf, DH * NH * DC);
    cvt(w_qr, wqr_bf, DRH * NH * DC);
    cvt(w_o, wo_bf, D * DH * NH);

    auto gx = [](int N) { return (N + BN - 1) / BN; };

    // c_kv[t,c] = sum_d h[t,d] w_dkv[c,d]
    gemm_nt<unsigned short, false><<<dim3(gx(DC), T / BM, 1), blk, 0, stream>>>(
        h_bf, D, 0, wdkv_bf, D, 0, ckv, DC, 1, 0, nullptr, 0, DC, D, 1.0f);
    // c_q
    gemm_nt<unsigned short, false><<<dim3(gx(DC), T / BM, 1), blk, 0, stream>>>(
        h_bf, D, 0, wdq_bf, D, 0, cq, DC, 1, 0, nullptr, 0, DC, D, 1.0f);

    // k_c per head -> kfull cols 0..127
    gemm_nt<unsigned short, false><<<dim3(gx(DH), T / BM, NH), blk, 0, stream>>>(
        ckv, DC, 0, wuk_bf, (long)NH * DC, DC,
        kfull, QD, 1, (long)T * QD, nullptr, 0, DH, DC, 1.0f);
    // v_c per head, stored transposed: vT[n][e][t]
    gemm_nt<unsigned short, false><<<dim3(gx(DH), T / BM, NH), blk, 0, stream>>>(
        ckv, DC, 0, wuv_bf, (long)NH * DC, DC,
        vT, 1, T, (long)DH * T, nullptr, 0, DH, DC, 1.0f);
    // q_c per head -> qfull cols 0..127
    gemm_nt<unsigned short, false><<<dim3(gx(DH), T / BM, NH), blk, 0, stream>>>(
        cq, DC, 0, wuq_bf, (long)NH * DC, DC,
        qfull, QD, 1, (long)T * QD, nullptr, 0, DH, DC, 1.0f);
    // q_r per head (pre-rope, fp32)
    gemm_nt<float, false><<<dim3(gx(DRH), T / BM, NH), blk, 0, stream>>>(
        cq, DC, 0, wqr_bf, (long)NH * DC, DC,
        qr_f32, DRH, 1, (long)T * DRH, nullptr, 0, DRH, DC, 1.0f);
    // k_r (pre-rope, fp32)
    gemm_nt<float, false><<<dim3(gx(DRH), T / BM, 1), blk, 0, stream>>>(
        h_bf, D, 0, wkr_bf, D, 0, kr_f32, DRH, 1, 0, nullptr, 0, DRH, D, 1.0f);

    rope_q_k<<<(NH * T * 32) / 256, 256, 0, stream>>>(qr_f32, freqs, qfull);
    rope_k_k<<<(T * 32) / 256, 256, 0, stream>>>(kr_f32, freqs, kfull);

    const float alpha_att = 1.0f / sqrtf((float)QD);
    for (int n = 0; n < NH; ++n) {
        const unsigned short* qn = qfull + (size_t)n * T * QD;
        const unsigned short* kn = kfull + (size_t)n * T * QD;
        const unsigned short* vn = vT + (size_t)n * DH * T;
        // logits[t,l] = q.k/sqrt(192) + mask[t,l]
        gemm_nt<float, true><<<dim3(gx(T), T / BM, 1), blk, 0, stream>>>(
            qn, QD, 0, kn, QD, 0, logits, T, 1, 0, mask, T, T, QD, alpha_att);
        softmax_k<<<T, 256, 0, stream>>>(logits, scoresT, T);
        // out[l,e] = sum_t sT[l,t] vT[e,t]  -> att[l, e*16+n]
        gemm_nt<unsigned short, false><<<dim3(gx(DH), T / BM, 1), blk, 0, stream>>>(
            scoresT, T, 0, vn, T, 0,
            att + n, (long)DH * NH, NH, 0, nullptr, 0, DH, T, 1.0f);
    }

    // O[l,d] = sum_k att[l,k] wo_bf[d,k]   (k = e*16+n matches w_o[d,e,n] layout)
    gemm_nt<float, false><<<dim3(gx(D), T / BM, 1), blk, 0, stream>>>(
        att, (long)DH * NH, 0, wo_bf, (long)DH * NH, 0,
        out, D, 1, 0, nullptr, 0, D, DH * NH, 1.0f);
}
